// GINEncoder_9251359555640
// MI455X (gfx1250) — matrix operand order repaired and checked by reading, not compiled, not run on hardware
//
#include <hip/hip_runtime.h>

// ---------------- problem constants (match reference) ----------------
#define NN   100000      // nodes
#define EE   640000      // edges
#define DD   128         // feature dim
#define LL   5           // layers
#define BN_EPS 1e-5f

typedef __attribute__((ext_vector_type(16))) _Float16 v16h;
typedef __attribute__((ext_vector_type(8)))  float    v8f;

// ---------------------------------------------------------------------
// Prep 1: convert + transpose weights to f16 [M,K] row-major.
//   Wt1[l][n<256][k<128] = W1[l][k][n]      (layer-1 weight, K=128, M=256)
//   Wt2[l][n<128][k<256] = W2[l][k][n]      (layer-2 weight, K=256, M=128)
__global__ void prep_weights(const float* __restrict__ W1,
                             const float* __restrict__ W2,
                             _Float16* __restrict__ Wt1,
                             _Float16* __restrict__ Wt2)
{
    int tid = blockIdx.x * blockDim.x + threadIdx.x;
    const int per_layer = 2 * DD * DD;            // 32768
    if (tid >= LL * per_layer) return;
    int l = tid / per_layer;
    int r = tid % per_layer;
    {   // Wt1: n in [0,256), k in [0,128)
        int n = r / DD, k = r % DD;
        Wt1[((size_t)l * 2 * DD + n) * DD + k] =
            (_Float16)W1[((size_t)l * DD + k) * 2 * DD + n];
    }
    {   // Wt2: n in [0,128), k in [0,256)
        int n = r / (2 * DD), k = r % (2 * DD);
        Wt2[((size_t)l * DD + n) * 2 * DD + k] =
            (_Float16)W2[((size_t)l * 2 * DD + k) * DD + n];
    }
}

// Prep 2: fold bias + BatchNorm into per-output-column scale/shift.
//   out = acc*sc + sh   where  sc = g*rsqrt(v+eps),  sh = (bias - m)*sc + beta
__global__ void prep_scales(const float* __restrict__ b1,
                            const float* __restrict__ g1,  const float* __restrict__ bb1,
                            const float* __restrict__ m1,  const float* __restrict__ v1,
                            const float* __restrict__ b2,
                            const float* __restrict__ go,  const float* __restrict__ bbo,
                            const float* __restrict__ mo,  const float* __restrict__ vo,
                            float* __restrict__ sc1, float* __restrict__ sh1,
                            float* __restrict__ sco, float* __restrict__ sho)
{
    int tid = blockIdx.x * blockDim.x + threadIdx.x;
    if (tid < LL * 2 * DD) {
        float s = g1[tid] * rsqrtf(v1[tid] + BN_EPS);
        sc1[tid] = s;
        sh1[tid] = (b1[tid] - m1[tid]) * s + bb1[tid];
    }
    if (tid < LL * DD) {
        float s = go[tid] * rsqrtf(vo[tid] + BN_EPS);
        sco[tid] = s;
        sho[tid] = (b2[tid] - mo[tid]) * s + bbo[tid];
    }
}

// ---------------------------------------------------------------------
// agg = h  (self term of GIN with eps=0), vectorized float4 copy
__global__ void copy_f4(const float4* __restrict__ src, float4* __restrict__ dst, int n4)
{
    int i = blockIdx.x * blockDim.x + threadIdx.x;
    if (i < n4) dst[i] = src[i];
}

// agg[dst[e]] += h[src[e]] : one wave per edge, float4 per lane,
// hardware global_atomic_add_f32 (no-return -> STOREcnt path).
__global__ void scatter_add(const float* __restrict__ h,
                            const int* __restrict__ ei,
                            float* __restrict__ agg)
{
    const int CH = DD / 4;                         // 32 float4 chunks per row
    long tid = (long)blockIdx.x * blockDim.x + threadIdx.x;
    long e = tid / CH;
    int  c = (int)(tid % CH);
    if (e >= EE) return;
    int s = ei[e];
    int d = ei[EE + e];
    float4 v = reinterpret_cast<const float4*>(h + (size_t)s * DD)[c];
    float* dp = agg + (size_t)d * DD + c * 4;
    unsafeAtomicAdd(dp + 0, v.x);
    unsafeAtomicAdd(dp + 1, v.y);
    unsafeAtomicAdd(dp + 2, v.z);
    unsafeAtomicAdd(dp + 3, v.w);
}

// ---------------------------------------------------------------------
// out[N,M] = act( (in[N,K] @ W[K,M]) * sc[col] + sh[col] )
//
// One wave32 owns a 16-row strip and FOUR consecutive 16-col tiles
// (a 16x64 output block): A is loaded + f32->f16 converted ONCE per
// 32-wide K-chunk and reused by 4 back-to-back v_wmma_f32_16x16x32_f16.
// Wt is f16, pre-transposed [M,K] row-major so B loads are contiguous b128s.
//
// A operand as loaded here (contiguous k, same order as B): lane holds row M=lane%16;
//   lanes 0-15 hold K = kb..kb+15, lanes 16-31 hold K = kb+16..kb+31.
// B operand as loaded here (same k order as A): lane holds col N=lane%16;
//   lanes 0-15 hold K = kb..kb+15, lanes 16-31 hold K = kb+16..kb+31.
// C/D f32: VGPR r -> row M = r + 8*(lane/16), col = lane%16.
template <int K, int M, bool RELU>
__global__ __launch_bounds__(256)
void gemm_bn_act(const float* __restrict__ in,
                 const _Float16* __restrict__ Wt,
                 const float* __restrict__ sc,
                 const float* __restrict__ sh,
                 float* __restrict__ out, int nWaves)
{
    int wave = blockIdx.x * (blockDim.x >> 5) + (threadIdx.x >> 5);
    if (wave >= nWaves) return;                    // wave-uniform: EXEC stays all-1s
    int lane = threadIdx.x & 31;
    const int NG = M / 64;                         // 16x64 blocks per row-strip
    int tm  = wave / NG;
    int tng = wave % NG;
    int rc  = lane & 15;                           // row (A) / col (B,D) within tile
    int hi  = lane >> 4;                           // half-wave selector

    const float* arow = in + (size_t)(tm * 16 + rc) * K;
    // 4 weight-panel base pointers (cols tng*64+rc, +16, +32, +48)
    const _Float16* bcol[4];
#pragma unroll
    for (int j = 0; j < 4; ++j)
        bcol[j] = Wt + (size_t)((tng * 4 + j) * 16 + rc) * K + hi * 16;

    v8f acc[4] = {v8f{}, v8f{}, v8f{}, v8f{}};

    for (int kb = 0; kb < K; kb += 32) {
        // ---- A: fp32 activations -> f16 operand (loaded once, used 4x) ----
        const float* g0 = arow + kb + hi * 16;
        const float* g1 = g0 + 8;
        v16h a;
#pragma unroll
        for (int i = 0; i < 8; ++i) {
            a[i]     = (_Float16)g0[i];
            a[i + 8] = (_Float16)g1[i];
        }
        // ---- 4 B panels, 4 WMMAs against the same A ----
#pragma unroll
        for (int j = 0; j < 4; ++j) {
            const _Float16* bp = bcol[j] + kb;
            v16h b;
#pragma unroll
            for (int i = 0; i < 16; ++i) b[i] = bp[i];
            acc[j] = __builtin_amdgcn_wmma_f32_16x16x32_f16(
                /*neg_a=*/false, a, /*neg_b=*/false, b,
                /*c_mod=*/(short)0, acc[j], /*reuse_a=*/false, /*reuse_b=*/false);
        }
    }

    // ---- epilogue: fused bias+BN (+ReLU), f32 stores ----
#pragma unroll
    for (int j = 0; j < 4; ++j) {
        int col = (tng * 4 + j) * 16 + rc;
        float s = sc[col];
        float t = sh[col];
#pragma unroll
        for (int r = 0; r < 8; ++r) {
            int orow = tm * 16 + r + hi * 8;
            float v = acc[j][r] * s + t;
            if (RELU) v = fmaxf(v, 0.0f);
            out[(size_t)orow * M + col] = v;
        }
    }
}

// ---------------------------------------------------------------------
extern "C" void kernel_launch(void* const* d_in, const int* in_sizes, int n_in,
                              void* d_out, int out_size, void* d_ws, size_t ws_size,
                              hipStream_t stream)
{
    const float* x     = (const float*)d_in[0];
    const int*   ei    = (const int*)  d_in[1];
    const float* W1    = (const float*)d_in[2];
    const float* b1    = (const float*)d_in[3];
    const float* bn1_g = (const float*)d_in[4];
    const float* bn1_b = (const float*)d_in[5];
    const float* bn1_m = (const float*)d_in[6];
    const float* bn1_v = (const float*)d_in[7];
    const float* W2    = (const float*)d_in[8];
    const float* b2    = (const float*)d_in[9];
    const float* bno_g = (const float*)d_in[10];
    const float* bno_b = (const float*)d_in[11];
    const float* bno_m = (const float*)d_in[12];
    const float* bno_v = (const float*)d_in[13];

    // -------- workspace carve-up --------
    char* ws = (char*)d_ws;
    size_t off = 0;
    float* P0 = (float*)(ws + off); off += (size_t)NN * DD * sizeof(float);        // h buffer
    float* P1 = (float*)(ws + off); off += (size_t)NN * DD * sizeof(float);        // agg buffer
    float* Z  = (float*)(ws + off); off += (size_t)NN * 2 * DD * sizeof(float);    // hidden
    _Float16* Wt1 = (_Float16*)(ws + off); off += (size_t)LL * 2 * DD * DD * sizeof(_Float16);
    _Float16* Wt2 = (_Float16*)(ws + off); off += (size_t)LL * 2 * DD * DD * sizeof(_Float16);
    float* sc1 = (float*)(ws + off); off += (size_t)LL * 2 * DD * sizeof(float);
    float* sh1 = (float*)(ws + off); off += (size_t)LL * 2 * DD * sizeof(float);
    float* sco = (float*)(ws + off); off += (size_t)LL * DD * sizeof(float);
    float* sho = (float*)(ws + off); off += (size_t)LL * DD * sizeof(float);

    // -------- prep (every call; deterministic) --------
    {
        int total = LL * 2 * DD * DD;                                  // 163840
        prep_weights<<<(total + 255) / 256, 256, 0, stream>>>(W1, W2, Wt1, Wt2);
        int sct = LL * 2 * DD;                                         // 1280
        prep_scales<<<(sct + 255) / 256, 256, 0, stream>>>(
            b1, bn1_g, bn1_b, bn1_m, bn1_v,
            b2, bno_g, bno_b, bno_m, bno_v,
            sc1, sh1, sco, sho);
    }

    const int n4        = NN * DD / 4;                 // 3.2M float4s
    const long scat_thr = (long)EE * (DD / 4);         // 20.48M threads
    const int  scat_blk = (int)((scat_thr + 255) / 256);
    const int  w1waves  = (NN / 16) * (2 * DD / 64);   // 25000 waves (M=256)
    const int  w2waves  = (NN / 16) * (DD / 64);       // 12500 waves (M=128)
    const int  w1blk    = (w1waves + 7) / 8;
    const int  w2blk    = (w2waves + 7) / 8;

    for (int l = 0; l < LL; ++l) {
        const float* h = (l == 0) ? x : P0;
        // agg = h + scatter(h[src] -> dst)
        copy_f4<<<(n4 + 255) / 256, 256, 0, stream>>>(
            (const float4*)h, (float4*)P1, n4);
        scatter_add<<<scat_blk, 256, 0, stream>>>(h, ei, P1);
        // z = relu(bn1(agg @ W1 + b1))
        gemm_bn_act<DD, 2 * DD, true><<<w1blk, 256, 0, stream>>>(
            P1, Wt1 + (size_t)l * 2 * DD * DD,
            sc1 + l * 2 * DD, sh1 + l * 2 * DD, Z, w1waves);
        // h' = bn_out(z @ W2 + b2), relu except last layer
        if (l != LL - 1) {
            gemm_bn_act<2 * DD, DD, true><<<w2blk, 256, 0, stream>>>(
                Z, Wt2 + (size_t)l * 2 * DD * DD,
                sco + l * DD, sho + l * DD, P0, w2waves);
        } else {
            gemm_bn_act<2 * DD, DD, false><<<w2blk, 256, 0, stream>>>(
                Z, Wt2 + (size_t)l * 2 * DD * DD,
                sco + l * DD, sho + l * DD, (float*)d_out, w2waves);
        }
    }
}
